// AttnBlock_83099027243617
// MI455X (gfx1250) — matrix operand order repaired and checked by reading, not compiled, not run on hardware
//
#include <hip/hip_runtime.h>

// ---------------------------------------------------------------------------
// AttnBlock for MI455X (gfx1250, wave32).
//   b=64 rows, c=2048 features.
//   GEMMs via v_wmma_f32_16x16x32_bf16 (memory-bound: 67MB of f32 weights,
//   ~2.9us at 23.3TB/s). Attention is exp2-bound (268M v_exp_f32); inner loop
//   packs the two per-thread softmax lanes into v_pk_fma_f32 so the plain
//   VALU side drops below the TRANS pipe (~1.5 VALU + 1 TRANS per element).
// ---------------------------------------------------------------------------

typedef __attribute__((ext_vector_type(16))) __bf16 v16bf;
typedef __attribute__((ext_vector_type(8)))  __bf16 v8bf;
typedef __attribute__((ext_vector_type(8)))  float  v8f;
typedef __attribute__((ext_vector_type(2)))  float  v2f;

#define C_DIM 2048
#define B_DIM 64

// ---------------------------------------------------------------------------
// Kernel 1: LayerNorm (weight only, bias=False), output bf16 for WMMA input.
// ---------------------------------------------------------------------------
__global__ __launch_bounds__(256) void ln_kernel(
    const float* __restrict__ x, const float* __restrict__ gamma,
    __bf16* __restrict__ hb)
{
    __shared__ float red[256];
    const int row = blockIdx.x;
    const int tid = threadIdx.x;
    const float* xr = x + row * C_DIM;

    float vals[8];
    float s = 0.f;
#pragma unroll
    for (int i = 0; i < 8; ++i) { vals[i] = xr[tid + i * 256]; s += vals[i]; }

    red[tid] = s; __syncthreads();
    for (int off = 128; off > 0; off >>= 1) {
        if (tid < off) red[tid] += red[tid + off];
        __syncthreads();
    }
    const float mu = red[0] * (1.f / C_DIM);
    __syncthreads();

    float vs = 0.f;
#pragma unroll
    for (int i = 0; i < 8; ++i) { float d = vals[i] - mu; vs += d * d; }
    red[tid] = vs; __syncthreads();
    for (int off = 128; off > 0; off >>= 1) {
        if (tid < off) red[tid] += red[tid + off];
        __syncthreads();
    }
    const float rinv = rsqrtf(red[0] * (1.f / C_DIM) + 1e-5f);

#pragma unroll
    for (int i = 0; i < 8; ++i) {
        const int c = tid + i * 256;
        hb[row * C_DIM + c] = (__bf16)((vals[i] - mu) * rinv * gamma[c]);
    }
}

// ---------------------------------------------------------------------------
// WMMA GEMM tile worker: C[m,n] = sum_k A[m,k] * W[n,k]  (+ optional residual)
// A: [64][2048] bf16 row-major.  W: [2048][2048] f32 row-major.
// Block = 256 threads = 8 waves; block covers M=64 x N=32, 64 K-steps of 32.
// f32 weights converted to bf16 inline (single f32 pass over HBM).
// ---------------------------------------------------------------------------
__device__ __forceinline__ void gemm_tile(
    const __bf16* __restrict__ A, const float* __restrict__ W,
    const float* __restrict__ resid, float* __restrict__ C, int nblk)
{
    const int lane  = threadIdx.x & 31;
    const int wave  = threadIdx.x >> 5;
    const int mBase = (wave & 3) * 16;
    const int nBase = nblk * 32 + (wave >> 2) * 16;

    // A frag (16-bit A 16x32, ISA 7.12.2): lane<16: M=lane, K={0..7,16..23};
    // lane>=16: M=lane-16, K={8..15,24..31}
    const int mrow = mBase + (lane & 15);
    const int aK0  = (lane >> 4) * 8;
    // B frag (16-bit B 32x16): lane<16: N=lane, K={0..7,16..23}; lane>=16: K={8..15,24..31}
    const int nrow = nBase + (lane & 15);
    const int bK0  = (lane >> 4) * 8;

    const __bf16* pa = A + mrow * C_DIM;
    const float*  pw = W + (size_t)nrow * C_DIM;

    v8f acc = {};
#pragma unroll 2
    for (int kb = 0; kb < C_DIM; kb += 32) {
        v8bf alo = *(const v8bf*)(pa + kb + aK0);
        v8bf ahi = *(const v8bf*)(pa + kb + aK0 + 16);
        v16bf a;
#pragma unroll
        for (int i = 0; i < 8; ++i) { a[i] = alo[i]; a[i + 8] = ahi[i]; }

        v16bf b;
#pragma unroll
        for (int i = 0; i < 16; ++i) b[i] = (__bf16)pw[kb + bK0 + (i & 7) + ((i >> 3) << 4)];

        acc = __builtin_amdgcn_wmma_f32_16x16x32_bf16(
            /*neg_a=*/false, a, /*neg_b=*/false, b,
            /*c_mod=*/(short)0, acc, /*reuse_a=*/false, /*reuse_b=*/false);
    }

    // D layout: acc[r] = (M = mBase + r + 8*(lane>=16), N = nBase + lane&15)
    const int mOut = mBase + ((lane >> 4) << 3);
    const int nOut = nBase + (lane & 15);
#pragma unroll
    for (int r = 0; r < 8; ++r) {
        const int row = mOut + r;
        float val = acc[r];
        if (resid) val += resid[row * C_DIM + nOut];
        C[row * C_DIM + nOut] = val;
    }
}

// q/k/v fused into one launch (grid = 3 * 64 blocks) for better WGP fill.
__global__ __launch_bounds__(256) void qkv_gemm_kernel(
    const __bf16* __restrict__ A,
    const float* __restrict__ W0, const float* __restrict__ W1,
    const float* __restrict__ W2, float* __restrict__ Cbase)
{
    const int mat  = blockIdx.x >> 6;
    const int nblk = blockIdx.x & 63;
    const float* W = (mat == 0) ? W0 : (mat == 1) ? W1 : W2;
    float* C = Cbase + mat * (B_DIM * C_DIM);
    gemm_tile(A, W, nullptr, C, nblk);
}

// Output projection + residual add.
__global__ __launch_bounds__(256) void out_gemm_kernel(
    const __bf16* __restrict__ A, const float* __restrict__ W,
    const float* __restrict__ resid, float* __restrict__ C)
{
    gemm_tile(A, W, resid, C, blockIdx.x);
}

// ---------------------------------------------------------------------------
// Kernel 3: rank-1 attention.  For each (b,p):
//   h2[b,p] = sum_j exp(q[b,p]*k[b,j]*scale) * v[b,j] / sum_j exp(...)
// Stable shift via per-batch kmax/kmin: max_j(alpha*kj) = alpha>0 ?
// alpha*kmax : alpha*kmin.  Work in log2 domain (one v_exp_f32 per element).
// Each thread handles TWO p positions so each LDS k/v fetch feeds two
// softmax lanes, and the two accumulator sets are kept in <2 x float>
// vectors to steer the backend into v_pk_fma_f32 packed math.
// ---------------------------------------------------------------------------
__global__ __launch_bounds__(256) void attn_kernel(
    const float* __restrict__ q, const float* __restrict__ k,
    const float* __restrict__ v, __bf16* __restrict__ h2b)
{
    __shared__ float ks[C_DIM];
    __shared__ float vs[C_DIM];
    __shared__ float rmax[256];
    __shared__ float rmin[256];

    const int b   = blockIdx.x >> 2;           // 4 blocks per batch row
    const int grp = blockIdx.x & 3;            // each covers 512 p positions
    const int tid = threadIdx.x;
    const int p0  = grp * 512 + tid;
    const int p1  = p0 + 256;

    // log2(e) / sqrt(2048)
    const float SC = 1.4426950408889634f / 45.254833995939045f;

    const float* kr = k + b * C_DIM;
    const float* vr = v + b * C_DIM;

    float lmax = -3.4e38f, lmin = 3.4e38f;
#pragma unroll
    for (int i = 0; i < 8; ++i) {
        const int j = tid + i * 256;
        const float kk = kr[j] * SC;
        ks[j] = kk;
        vs[j] = vr[j];
        lmax = fmaxf(lmax, kk);
        lmin = fminf(lmin, kk);
    }
    rmax[tid] = lmax; rmin[tid] = lmin;
    __syncthreads();
    for (int off = 128; off > 0; off >>= 1) {
        if (tid < off) {
            rmax[tid] = fmaxf(rmax[tid], rmax[tid + off]);
            rmin[tid] = fminf(rmin[tid], rmin[tid + off]);
        }
        __syncthreads();
    }
    const float kmax = rmax[0];
    const float kmin = rmin[0];

    const float a0 = q[b * C_DIM + p0];
    const float a1 = q[b * C_DIM + p1];
    const v2f alpha2 = {a0, a1};
    const v2f negm2  = {-((a0 > 0.f) ? a0 * kmax : a0 * kmin),
                        -((a1 > 0.f) ? a1 * kmax : a1 * kmin)};

    v2f num2 = {0.f, 0.f};
    v2f den2 = {0.f, 0.f};
    for (int j = 0; j < C_DIM; j += 4) {
        const float4 k4 = *(const float4*)(ks + j);
        const float4 v4 = *(const float4*)(vs + j);
#pragma unroll
        for (int u = 0; u < 4; ++u) {
            const float kj = (&k4.x)[u];
            const float vj = (&v4.x)[u];
            const v2f kv2 = {kj, kj};
            const v2f vv2 = {vj, vj};
            const v2f arg = __builtin_elementwise_fma(alpha2, kv2, negm2);
            const v2f ex  = {__builtin_amdgcn_exp2f(arg.x),
                             __builtin_amdgcn_exp2f(arg.y)};
            num2 = __builtin_elementwise_fma(ex, vv2, num2);
            den2 += ex;
        }
    }
    h2b[b * C_DIM + p0] = (__bf16)(num2.x / den2.x);
    h2b[b * C_DIM + p1] = (__bf16)(num2.y / den2.y);
}

// ---------------------------------------------------------------------------
// Launch
// ---------------------------------------------------------------------------
extern "C" void kernel_launch(void* const* d_in, const int* in_sizes, int n_in,
                              void* d_out, int out_size, void* d_ws, size_t ws_size,
                              hipStream_t stream) {
    const float* x     = (const float*)d_in[0];
    const float* gamma = (const float*)d_in[1];
    const float* Wq    = (const float*)d_in[2];
    const float* Wk    = (const float*)d_in[3];
    const float* Wv    = (const float*)d_in[4];
    const float* Wo    = (const float*)d_in[5];
    float* out = (float*)d_out;

    char* ws = (char*)d_ws;
    const size_t N_ELEM = (size_t)B_DIM * C_DIM;         // 131072
    __bf16* hb   = (__bf16*)ws;                          // 256 KB
    float*  qkv  = (float*)(ws + N_ELEM * 2);            // 3 x 512 KB (q,k,v)
    float*  qbuf = qkv;
    float*  kbuf = qkv + N_ELEM;
    float*  vbuf = qkv + 2 * N_ELEM;
    __bf16* h2b  = (__bf16*)(ws + N_ELEM * 2 + 3 * N_ELEM * 4);  // 256 KB

    ln_kernel<<<B_DIM, 256, 0, stream>>>(x, gamma, hb);
    qkv_gemm_kernel<<<3 * 64, 256, 0, stream>>>(hb, Wq, Wk, Wv, qkv);
    attn_kernel<<<B_DIM * 4, 256, 0, stream>>>(qbuf, kbuf, vbuf, h2b);
    out_gemm_kernel<<<64, 256, 0, stream>>>(h2b, Wo, x, out);
}
